// VideoChapterWindowAttention_66159676227617
// MI455X (gfx1250) — matrix operand order repaired and checked by reading, not compiled, not run on hardware
//
#include <hip/hip_runtime.h>
#include <hip/hip_bf16.h>

#define B_   2048
#define S_   257
#define H_   768
#define NH_  12
#define HD_  64
#define MID_ 128
#define SP_  288   // window padded to 9*32 for K-steps

typedef _Float16 h8  __attribute__((ext_vector_type(8)));
typedef _Float16 h16 __attribute__((ext_vector_type(16)));
typedef float    f8  __attribute__((ext_vector_type(8)));
typedef unsigned int uv4 __attribute__((ext_vector_type(4)));
typedef int          iv4 __attribute__((ext_vector_type(4)));
typedef int          iv8 __attribute__((ext_vector_type(8)));

__device__ __forceinline__ h16 mk16(h8 lo, h8 hi) {
  h16 r;
#pragma unroll
  for (int i = 0; i < 8; ++i) { r[i] = lo[i]; r[8 + i] = hi[i]; }
  return r;
}

// ---- A operand (16x32 f16, MxK): lane holds 16 consecutive K at 16*(lane/16) ----
__device__ __forceinline__ h16 aload_f16(const _Float16* p) {
  h8 lo = *(const h8*)p;
  h8 hi = *(const h8*)(p + 8);
  return mk16(lo, hi);
}
__device__ __forceinline__ h16 aload_f32(const float* p) {
  h16 r;
#pragma unroll
  for (int i = 0; i < 8; ++i) { r[i] = (_Float16)p[i]; r[8 + i] = (_Float16)p[8 + i]; }
  return r;
}
__device__ __forceinline__ h16 aload_f32_pe(const float* p, const float* q) {
  h16 r;
#pragma unroll
  for (int i = 0; i < 8; ++i) {
    r[i]     = (_Float16)(p[i]      + q[i]);
    r[8 + i] = (_Float16)(p[8 + i] + q[8 + i]);
  }
  return r;
}

// ---- B operand (32x16 f16, KxN): lane holds 16 consecutive K at 16*(lane/16) ----
__device__ __forceinline__ h16 bload_f16(const _Float16* p) {
  h8 lo = *(const h8*)p;
  h8 hi = *(const h8*)(p + 8);
  return mk16(lo, hi);
}
__device__ __forceinline__ h16 bload_f32(const float* p) {
  h16 r;
#pragma unroll
  for (int i = 0; i < 16; ++i) r[i] = (_Float16)p[i];
  return r;
}
__device__ __forceinline__ h16 bload_f32_pe(const float* p, const float* q) {
  h16 r;
#pragma unroll
  for (int i = 0; i < 16; ++i) r[i] = (_Float16)(p[i] + q[i]);
  return r;
}

__device__ __forceinline__ f8 wmma_f16(h16 a, h16 b, f8 c) {
  return __builtin_amdgcn_wmma_f32_16x16x32_f16(false, a, false, b, (short)0, c,
                                                false, false);
}

// ============================================================================
// Tensor Data Mover: 2D tile load Global -> LDS (D# per CDNA5 ISA ch.8).
//   data_size = 4B, tile = tile_w x tile_h, row stride = stride elems,
//   rows >= tensor_h are zero-filled by the TDM (OOB reads return 0).
// ============================================================================
#if __has_builtin(__builtin_amdgcn_tensor_load_to_lds)
#define USE_TDM 1
__device__ __forceinline__ void tdm_load_2d(unsigned lds_off, const void* gptr,
                                            unsigned tile_w, unsigned tile_h,
                                            unsigned tensor_w, unsigned tensor_h,
                                            unsigned stride_elems) {
  unsigned long long ga = (unsigned long long)gptr;
  uv4 g0;
  g0[0] = 1u;                                        // count=1, user descriptor
  g0[1] = lds_off;                                   // lds_addr (bytes)
  g0[2] = (unsigned)(ga & 0xffffffffu);              // global_addr[31:0]
  g0[3] = (unsigned)((ga >> 32) & 0x01ffffffu)       // global_addr[56:32]
          | (2u << 30);                              // type = 2 ("image")
  iv8 g1;
  g1[0] = (int)(2u << 16);                           // wg_mask=0, data_size=2 (4B)
  g1[1] = (int)((tensor_w & 0xffffu) << 16);         // tensor_dim0[15:0]
  g1[2] = (int)(((tensor_w >> 16) & 0xffffu) |       // tensor_dim0[31:16]
                ((tensor_h & 0xffffu) << 16));       // tensor_dim1[15:0]
  g1[3] = (int)(((tensor_h >> 16) & 0xffffu) |       // tensor_dim1[31:16]
                ((tile_w & 0xffffu) << 16));         // tile_dim0
  g1[4] = (int)(tile_h & 0xffffu);                   // tile_dim1 (tile_dim2=0)
  g1[5] = (int)stride_elems;                         // tensor_dim0_stride[31:0]
  g1[6] = 0;                                         // stride hi, dim1_stride lo
  g1[7] = 0;
  iv4 z4 = {0, 0, 0, 0};
#if __clang_major__ >= 23
  iv8 z8 = {0, 0, 0, 0, 0, 0, 0, 0};
  __builtin_amdgcn_tensor_load_to_lds(g0, g1, z4, z4, z8, 0);
#else
  __builtin_amdgcn_tensor_load_to_lds(g0, g1, z4, z4, 0);
#endif
}
#else
#define USE_TDM 0
#endif

// ============================================================================
// K1: pe[s, j] = LayerNorm(pos[s] * pe_w[j] + pe_b[j]) * ln_g + ln_b   (f32)
// ============================================================================
__global__ __launch_bounds__(256) void pe_kernel(const float* __restrict__ pe_w,
                                                 const float* __restrict__ pe_b,
                                                 const float* __restrict__ ln_g,
                                                 const float* __restrict__ ln_b,
                                                 float* __restrict__ pe) {
  __shared__ float red[256];
  const int s = blockIdx.x, t = threadIdx.x;
  const float pos = ((float)s - (float)MID_) / ((float)MID_ + 1e-6f);
  float x[3], sum = 0.f;
#pragma unroll
  for (int i = 0; i < 3; ++i) {
    int j = t + 256 * i;
    x[i] = pos * pe_w[j] + pe_b[j];
    sum += x[i];
  }
  red[t] = sum; __syncthreads();
  for (int o = 128; o > 0; o >>= 1) { if (t < o) red[t] += red[t + o]; __syncthreads(); }
  const float mu = red[0] / (float)H_;
  __syncthreads();
  float vs = 0.f;
#pragma unroll
  for (int i = 0; i < 3; ++i) { float d = x[i] - mu; vs += d * d; }
  red[t] = vs; __syncthreads();
  for (int o = 128; o > 0; o >>= 1) { if (t < o) red[t] += red[t + o]; __syncthreads(); }
  const float rstd = rsqrtf(red[0] / (float)H_ + 1e-5f);
#pragma unroll
  for (int i = 0; i < 3; ++i) {
    int j = t + 256 * i;
    pe[(size_t)s * H_ + j] = (x[i] - mu) * rstd * ln_g[j] + ln_b[j];
  }
}

// ============================================================================
// K2: wkT[n, m] = (f16) Wk[m, n]
// ============================================================================
__global__ __launch_bounds__(256) void wkT_kernel(const float* __restrict__ Wk,
                                                  _Float16* __restrict__ wkT) {
  int i = blockIdx.x * 256 + threadIdx.x;   // over H_*H_
  int m = i / H_, n = i % H_;
  wkT[(size_t)n * H_ + m] = (_Float16)Wk[i];
}

// ============================================================================
// K3: q = (hidden[:,mid,:] + pe[mid]) @ Wq^T + bq      [B, H] f32, WMMA
// ============================================================================
__global__ __launch_bounds__(32) void q_kernel(const float* __restrict__ hidden,
                                               const float* __restrict__ pe,
                                               const float* __restrict__ Wq,
                                               const float* __restrict__ bq,
                                               float* __restrict__ q) {
  const int bt = blockIdx.x, nt = blockIdx.y;
  const int lane = threadIdx.x, col = lane & 15;
  const int akh = (lane >> 4) << 4, bkh = (lane >> 4) << 4;
  const float* arow = hidden + ((size_t)(bt * 16 + col) * S_ + MID_) * H_;
  const float* prow = pe + (size_t)MID_ * H_;
  const float* brow = Wq + (size_t)(nt * 16 + col) * H_;   // B[k,n] = Wq[n,k]
  f8 c = {0.f, 0.f, 0.f, 0.f, 0.f, 0.f, 0.f, 0.f};
  for (int ks = 0; ks < H_; ks += 32) {
    h16 a = aload_f32_pe(arow + ks + akh, prow + ks + akh);
    h16 b = bload_f32(brow + ks + bkh);
    c = wmma_f16(a, b, c);
  }
  const float bias = bq[nt * 16 + col];
#pragma unroll
  for (int r = 0; r < 8; ++r) {
    int m = r + ((lane >> 4) << 3);
    q[(size_t)(bt * 16 + m) * H_ + nt * 16 + col] = c[r] + bias;
  }
}

// ============================================================================
// K4: qk[b, h, :] = 0.125 * q[b, h*64:+64] @ Wk[h*64:+64, :]   f16, WMMA
// ============================================================================
__global__ __launch_bounds__(32) void qk_kernel(const float* __restrict__ q,
                                                const _Float16* __restrict__ wkT,
                                                _Float16* __restrict__ qk) {
  const int bt = blockIdx.x, h = blockIdx.y, jt = blockIdx.z;
  const int lane = threadIdx.x, col = lane & 15;
  const int akh = (lane >> 4) << 4, bkh = (lane >> 4) << 4;
  const float*    arow = q + (size_t)(bt * 16 + col) * H_ + h * HD_;
  const _Float16* brow = wkT + (size_t)(jt * 16 + col) * H_ + h * HD_;
  f8 c = {0.f, 0.f, 0.f, 0.f, 0.f, 0.f, 0.f, 0.f};
#pragma unroll
  for (int ks = 0; ks < HD_; ks += 32) {
    h16 a = aload_f32(arow + ks + akh);
    h16 b = bload_f16(brow + ks + bkh);
    c = wmma_f16(a, b, c);
  }
#pragma unroll
  for (int r = 0; r < 8; ++r) {
    int m = r + ((lane >> 4) << 3);   // batch row within tile
    qk[((size_t)(bt * 16 + m) * 16 + h) * H_ + jt * 16 + col] =
        (_Float16)(c[r] * 0.125f);    // fold 1/sqrt(HEAD_DIM)
  }
}

// ============================================================================
// K5 (heavy stream #1): scores[b, h, s] = (hidden[b,s]+pe[s]) . qk[b,h]
// One wave computes FOUR s-tiles, sharing each A (qk) load across 4 WMMAs.
// grid (B, 5) covers ceil(S/16)=17 tiles (extra tiles clamp & skip store).
// ============================================================================
__global__ __launch_bounds__(32) void scores_kernel(const float* __restrict__ hidden,
                                                    const float* __restrict__ pe,
                                                    const _Float16* __restrict__ qk,
                                                    float* __restrict__ scores) {
  const int b = blockIdx.x, sg = blockIdx.y;
  const int lane = threadIdx.x, col = lane & 15;
  const int akh = (lane >> 4) << 4, bkh = (lane >> 4) << 4;
  const _Float16* arow = qk + (size_t)b * 16 * H_ + col * H_;   // row = head
  const float* hrow[4];
  const float* prow[4];
#pragma unroll
  for (int t = 0; t < 4; ++t) {
    const int s = min((sg * 4 + t) * 16 + col, S_ - 1);
    hrow[t] = hidden + ((size_t)b * S_ + s) * H_;
    prow[t] = pe + (size_t)s * H_;
  }
  f8 c[4];
#pragma unroll
  for (int t = 0; t < 4; ++t) c[t] = (f8){0.f, 0.f, 0.f, 0.f, 0.f, 0.f, 0.f, 0.f};
  for (int ks = 0; ks < H_; ks += 32) {
    h16 a = aload_f16(arow + ks + akh);
#pragma unroll
    for (int t = 0; t < 4; ++t) {
      __builtin_prefetch(hrow[t] + ks + 128, 0, 1);   // global_prefetch_b8
      h16 bb = bload_f32_pe(hrow[t] + ks + bkh, prow[t] + ks + bkh);
      c[t] = wmma_f16(a, bb, c[t]);
    }
  }
#pragma unroll
  for (int t = 0; t < 4; ++t) {
    const int sv = (sg * 4 + t) * 16 + col;
    if (sv < S_) {
#pragma unroll
      for (int r = 0; r < 8; ++r) {
        int m = r + ((lane >> 4) << 3);           // head
        if (m < NH_)
          scores[((size_t)b * NH_ + m) * S_ + sv] = c[t][r];
      }
    }
  }
}

// ============================================================================
// K6: softmax over s with additive pos_bias; probs f16 [B,16,SP_], K-padded
//     with zeros and head rows 12..15 zeroed (so the u-pass A operand is clean).
// ============================================================================
__global__ __launch_bounds__(32) void softmax_kernel(const float* __restrict__ scores,
                                                     const float* __restrict__ pos_bias,
                                                     _Float16* __restrict__ probs) {
  const int b = blockIdx.x, h = blockIdx.y, t = threadIdx.x;
  _Float16* prow = probs + ((size_t)b * 16 + h) * SP_;
  if (h >= NH_) {
    for (int s = t; s < SP_; s += 32) prow[s] = (_Float16)0.f;
    return;
  }
  const float* srow = scores + ((size_t)b * NH_ + h) * S_;
  const float* pb = pos_bias + (size_t)h * S_;
  float x[SP_ / 32];
  float mx = -1e30f;
#pragma unroll
  for (int i = 0; i < SP_ / 32; ++i) {
    int s = t + 32 * i;
    float v = (s < S_) ? (srow[s] + pb[s]) : -1e30f;
    x[i] = v;
    mx = fmaxf(mx, v);
  }
  for (int o = 16; o > 0; o >>= 1) mx = fmaxf(mx, __shfl_xor(mx, o, 32));
  float sum = 0.f;
#pragma unroll
  for (int i = 0; i < SP_ / 32; ++i) {
    float e = (x[i] > -1e29f) ? __expf(x[i] - mx) : 0.f;
    x[i] = e;
    sum += e;
  }
  for (int o = 16; o > 0; o >>= 1) sum += __shfl_xor(sum, o, 32);
  const float inv = 1.f / sum;
#pragma unroll
  for (int i = 0; i < SP_ / 32; ++i) prow[t + 32 * i] = (_Float16)(x[i] * inv);
}

// ============================================================================
// K7 (heavy stream #2): u[b, h, j] = sum_s probs[b,h,s] * (hidden[b,s,j]+pe[s,j])
// One wave owns a 64-wide j-slab: per 32-row k-step the TDM DMAs the 32x64
// hidden tile and the matching pe tile into LDS (TENSORcnt-synchronized),
// then 4 B-operands are gathered transposed and share one A (probs) load.
// Fallback path stages through LDS with plain loads if the TDM builtin is
// unavailable. TDM zero-fills rows >= S (probs K-pad is zero anyway).
// ============================================================================
__global__ __launch_bounds__(32) void u_kernel(const float* __restrict__ hidden,
                                               const float* __restrict__ pe,
                                               const _Float16* __restrict__ probs,
                                               _Float16* __restrict__ u) {
  __shared__ float hs_t[32 * 64];   // [k-row][64 j]
  __shared__ float pe_t[32 * 64];
  const int b = blockIdx.x, jb = blockIdx.y;
  const int j0 = jb * 64;
  const int lane = threadIdx.x, col = lane & 15;
  const int akh = (lane >> 4) << 4, bkh = (lane >> 4) << 4;
  const _Float16* arow = probs + ((size_t)b * 16 + col) * SP_;
  f8 c[4];
#pragma unroll
  for (int t = 0; t < 4; ++t) c[t] = (f8){0.f, 0.f, 0.f, 0.f, 0.f, 0.f, 0.f, 0.f};
#if USE_TDM
  const unsigned hs_off = (unsigned)(uintptr_t)(void*)&hs_t[0];  // LDS byte offset
  const unsigned pe_off = (unsigned)(uintptr_t)(void*)&pe_t[0];
#endif
  for (int ks = 0; ks < SP_; ks += 32) {
    __syncthreads();   // previous gathers complete before LDS overwrite
#if USE_TDM
    const unsigned rows = (unsigned)((S_ - ks) < 32 ? (S_ - ks) : 32);
    tdm_load_2d(hs_off, hidden + ((size_t)b * S_ + ks) * H_ + j0, 64u, 32u,
                64u, rows, (unsigned)H_);
    tdm_load_2d(pe_off, pe + (size_t)ks * H_ + j0, 64u, 32u,
                64u, rows, (unsigned)H_);
    __builtin_amdgcn_s_wait_tensorcnt(0);
#else
    const int s = min(ks + lane, S_ - 1);
    const float* hr = hidden + ((size_t)b * S_ + s) * H_ + j0;
    const float* pr = pe + (size_t)s * H_ + j0;
#pragma unroll
    for (int i = 0; i < 64; ++i) {
      hs_t[lane * 64 + i] = hr[i];
      pe_t[lane * 64 + i] = pr[i];
    }
    __syncthreads();
#endif
    h16 a = aload_f16(arow + ks + akh);
#pragma unroll
    for (int t = 0; t < 4; ++t) {
      h16 bb;
#pragma unroll
      for (int i = 0; i < 16; ++i) {   // transpose gather from LDS
        const int k = bkh + i;
        bb[i] = (_Float16)(hs_t[k * 64 + t * 16 + col] + pe_t[k * 64 + t * 16 + col]);
      }
      c[t] = wmma_f16(a, bb, c[t]);
    }
  }
#pragma unroll
  for (int t = 0; t < 4; ++t)
#pragma unroll
    for (int r = 0; r < 8; ++r) {
      int m = r + ((lane >> 4) << 3);   // head
      u[((size_t)b * 16 + m) * H_ + j0 + t * 16 + col] = (_Float16)c[t][r];
    }
}

// ============================================================================
// K8: ctx[b, h*64+d] = u[b,h,:] . Wv[h*64+d, :] + bv[h*64+d]     WMMA
// ============================================================================
__global__ __launch_bounds__(32) void ctx_kernel(const _Float16* __restrict__ u,
                                                 const float* __restrict__ Wv,
                                                 const float* __restrict__ bv,
                                                 float* __restrict__ ctx) {
  const int bt = blockIdx.x, h = blockIdx.y, dt = blockIdx.z;
  const int lane = threadIdx.x, col = lane & 15;
  const int akh = (lane >> 4) << 4, bkh = (lane >> 4) << 4;
  const _Float16* arow = u + ((size_t)(bt * 16 + col) * 16 + h) * H_;
  const float* brow = Wv + (size_t)(h * HD_ + dt * 16 + col) * H_;  // B[k,n]=Wv[h*64+n,k]
  f8 c = {0.f, 0.f, 0.f, 0.f, 0.f, 0.f, 0.f, 0.f};
  for (int ks = 0; ks < H_; ks += 32) {
    h16 a = aload_f16(arow + ks + akh);
    h16 b = bload_f32(brow + ks + bkh);
    c = wmma_f16(a, b, c);
  }
  const float bias = bv[h * HD_ + dt * 16 + col];
#pragma unroll
  for (int r = 0; r < 8; ++r) {
    int m = r + ((lane >> 4) << 3);
    ctx[(size_t)(bt * 16 + m) * H_ + h * HD_ + dt * 16 + col] = c[r] + bias;
  }
}

// ============================================================================
// K9: out = ctx @ Wo^T + bo      [B, 1, H] f32, WMMA
// ============================================================================
__global__ __launch_bounds__(32) void out_kernel(const float* __restrict__ ctx,
                                                 const float* __restrict__ Wo,
                                                 const float* __restrict__ bo,
                                                 float* __restrict__ out) {
  const int bt = blockIdx.x, nt = blockIdx.y;
  const int lane = threadIdx.x, col = lane & 15;
  const int akh = (lane >> 4) << 4, bkh = (lane >> 4) << 4;
  const float* arow = ctx + (size_t)(bt * 16 + col) * H_;
  const float* brow = Wo + (size_t)(nt * 16 + col) * H_;   // B[k,n] = Wo[n,k]
  f8 c = {0.f, 0.f, 0.f, 0.f, 0.f, 0.f, 0.f, 0.f};
  for (int ks = 0; ks < H_; ks += 32) {
    h16 a = aload_f32(arow + ks + akh);
    h16 b = bload_f32(brow + ks + bkh);
    c = wmma_f16(a, b, c);
  }
  const float bias = bo[nt * 16 + col];
#pragma unroll
  for (int r = 0; r < 8; ++r) {
    int m = r + ((lane >> 4) << 3);
    out[(size_t)(bt * 16 + m) * H_ + nt * 16 + col] = c[r] + bias;
  }
}

extern "C" void kernel_launch(void* const* d_in, const int* in_sizes, int n_in,
                              void* d_out, int out_size, void* d_ws, size_t ws_size,
                              hipStream_t stream) {
  (void)in_sizes; (void)n_in; (void)out_size; (void)ws_size;
  const float* hidden   = (const float*)d_in[0];
  const float* Wq       = (const float*)d_in[1];
  const float* bq       = (const float*)d_in[2];
  const float* Wk       = (const float*)d_in[3];
  // d_in[4] = bk: a per-(b,h) constant over s -> cancels in softmax, unused.
  const float* Wv       = (const float*)d_in[5];
  const float* bv       = (const float*)d_in[6];
  const float* Wo       = (const float*)d_in[7];
  const float* bo       = (const float*)d_in[8];
  const float* pe_w     = (const float*)d_in[9];
  const float* pe_b     = (const float*)d_in[10];
  const float* ln_g     = (const float*)d_in[11];
  const float* ln_b     = (const float*)d_in[12];
  const float* pos_bias = (const float*)d_in[13];
  float* out = (float*)d_out;

  char* w = (char*)d_ws;
  size_t off = 0;
  auto take = [&](size_t bytes) -> char* {
    char* p = w + off;
    off = (off + bytes + 255) & ~(size_t)255;
    return p;
  };
  float*    pe    = (float*)    take((size_t)S_ * H_ * 4);        // ~0.79 MB
  float*    qbuf  = (float*)    take((size_t)B_ * H_ * 4);        //  6.3 MB
  _Float16* wkT   = (_Float16*) take((size_t)H_ * H_ * 2);        //  1.2 MB
  _Float16* qk    = (_Float16*) take((size_t)B_ * 16 * H_ * 2);   // 50.3 MB
  float*    sc    = (float*)    take((size_t)B_ * NH_ * S_ * 4);  // 25.3 MB
  _Float16* probs = (_Float16*) take((size_t)B_ * 16 * SP_ * 2);  // 18.9 MB
  _Float16* ubuf  = (_Float16*) take((size_t)B_ * 16 * H_ * 2);   // 50.3 MB
  float*    ctx   = (float*)    take((size_t)B_ * H_ * 4);        //  6.3 MB

  pe_kernel<<<S_, 256, 0, stream>>>(pe_w, pe_b, ln_g, ln_b, pe);
  wkT_kernel<<<(H_ * H_) / 256, 256, 0, stream>>>(Wk, wkT);
  q_kernel<<<dim3(B_ / 16, H_ / 16), 32, 0, stream>>>(hidden, pe, Wq, bq, qbuf);
  qk_kernel<<<dim3(B_ / 16, NH_, H_ / 16), 32, 0, stream>>>(qbuf, wkT, qk);
  scores_kernel<<<dim3(B_, 5), 32, 0, stream>>>(hidden, pe, qk, sc);
  softmax_kernel<<<dim3(B_, 16), 32, 0, stream>>>(sc, pos_bias, probs);
  u_kernel<<<dim3(B_, H_ / 64), 32, 0, stream>>>(hidden, pe, probs, ubuf);
  ctx_kernel<<<dim3(B_ / 16, NH_, HD_ / 16), 32, 0, stream>>>(ubuf, Wv, bv, ctx);
  out_kernel<<<dim3(B_ / 16, H_ / 16), 32, 0, stream>>>(ctx, Wo, bo, out);
}
